// RGCN_13589276524585
// MI455X (gfx1250) — compile-verified, matrix operand order checked by reading, not run on hardware
//
#include <hip/hip_runtime.h>
#include <hip/hip_bf16.h>

typedef __attribute__((ext_vector_type(16))) __bf16       v16bf;
typedef __attribute__((ext_vector_type(8)))  float        v8f;
typedef __attribute__((ext_vector_type(4)))  unsigned int v4u;
typedef __attribute__((ext_vector_type(8)))  int          v8i;
typedef __attribute__((ext_vector_type(4)))  int          v4i;

#define DIM     128
#define NBASES  16
#define NREL    100
#define CHUNK   256          // edges per block (16 tiles of 16 edges, 8 waves x 2)
#define WSTRIDE (DIM + 16)   // padded LDS row stride (bf16 elems): 288B, 32B-aligned rows

#if defined(__has_builtin)
#  if __has_builtin(__builtin_amdgcn_tensor_load_to_lds)
#    define HAVE_TDM 1
#  else
#    define HAVE_TDM 0
#  endif
#else
#  define HAVE_TDM 0
#endif

// ---------------------------------------------------------------------------
// TDM: DMA one 128x128 bf16 tile (row-major, 256B rows) into LDS with 32B row
// padding (-> 288B LDS stride == WSTRIDE). D# per CDNA5 ISA ch.8:
//   pad_interval code 5 = 64 DWORDs (256B), pad_amount code 7 = 8 DWORDs (32B)
// ---------------------------------------------------------------------------
__device__ __forceinline__ void tdm_load_tile(const __bf16* gsrc, unsigned ldsOff) {
#if HAVE_TDM
    unsigned long long ga = (unsigned long long)(uintptr_t)gsrc;
    v4u g0;
    g0[0] = 1u;                                   // count=1 (valid descriptor)
    g0[1] = ldsOff;                               // lds_addr  (bits 63:32)
    g0[2] = (unsigned)(ga & 0xFFFFFFFFu);         // global_addr low (bits 95:64)
    g0[3] = (unsigned)((ga >> 32) & 0x01FFFFFFu)  // global_addr high (57-bit)
          | (2u << 30);                           // type=2 ("image") bits 127:126
    v8i g1;
    g1[0] = (1 << 16)                             // data_size=1 -> 2 bytes
          | (1 << 20)                             // pad_enable
          | (5 << 22)                             // pad_interval: 64 DWORDs
          | (7 << 25);                            // pad_amount: 8 DWORDs
    g1[1] = (int)(128u << 16);                    // tensor_dim0=128 (low16 @bit48)
    g1[2] = (int)(128u << 16);                    // tensor_dim1=128 (low16 @bit80)
    g1[3] = (int)(128u << 16);                    // tile_dim0=128   (@bits 127:112)
    g1[4] = 128;                                  // tile_dim1=128   (@bits 143:128)
    g1[5] = 128;                                  // tensor_dim0_stride=128 (@bit160)
    g1[6] = 0;
    g1[7] = 0;
    v4i z4 = {0, 0, 0, 0};
#if __clang_major__ >= 23
    v8i z8 = {0, 0, 0, 0, 0, 0, 0, 0};
    __builtin_amdgcn_tensor_load_to_lds(g0, g1, z4, z4, z8, 0);
#else
    __builtin_amdgcn_tensor_load_to_lds(g0, g1, z4, z4, 0);
#endif
    __builtin_amdgcn_s_wait_tensorcnt(0);
#endif
}

// fallback cooperative stage (also used when TDM builtin unavailable)
__device__ __forceinline__ void coop_load_tile(const __bf16* gsrc, __bf16* sDst) {
#if !HAVE_TDM
    for (int i = threadIdx.x; i < DIM * (DIM / 8); i += blockDim.x) {
        int row = i >> 4, c16 = i & 15;   // 16 uint4 per 128-elem row
        ((uint4*)&sDst[row * WSTRIDE])[c16] = ((const uint4*)(gsrc + row * DIM))[c16];
    }
#endif
}

// ---------------------------------------------------------------------------
// W_r[n][k] = sum_b att[r,b] * basis[b][k][n]   (transposed, bf16, per relation)
// ---------------------------------------------------------------------------
__global__ void build_w(const float* __restrict__ basis, const float* __restrict__ att,
                        __bf16* __restrict__ Wt) {
    __shared__ float sAtt[NBASES];
    int r = blockIdx.x;
    if (threadIdx.x < NBASES) sAtt[threadIdx.x] = att[r * NBASES + threadIdx.x];
    __syncthreads();
    for (int f = threadIdx.x; f < DIM * DIM; f += blockDim.x) {
        int n = f >> 7, k = f & 127;
        float acc = 0.f;
#pragma unroll
        for (int b = 0; b < NBASES; ++b)
            acc += sAtt[b] * basis[b * DIM * DIM + k * DIM + n];
        Wt[(size_t)r * DIM * DIM + f] = (__bf16)acc;
    }
}

// rootT[n][k] = root[k][n], bf16
__global__ void build_rt(const float* __restrict__ root, __bf16* __restrict__ rt) {
    int f = blockIdx.x * blockDim.x + threadIdx.x;
    if (f < DIM * DIM) { int n = f >> 7, k = f & 127; rt[f] = (__bf16)root[k * DIM + n]; }
}

// xb[i][j] = bf16(emb[entity[i]][j])
__global__ void gather_cast(const int* __restrict__ entity, const float* __restrict__ emb,
                            __bf16* __restrict__ xb, int total) {
    int i = blockIdx.x * blockDim.x + threadIdx.x;
    if (i < total) {
        int node = i >> 7, j = i & 127;
        xb[i] = (__bf16)emb[(size_t)entity[node] * DIM + j];
    }
}

// per-relation histogram + per-dst in-degree (in-degree reused by both layers)
__global__ void hist(const int* __restrict__ etype, const int* __restrict__ dstA,
                     int* __restrict__ relCount, float* __restrict__ cnt, int E) {
    int e = blockIdx.x * blockDim.x + threadIdx.x;
    if (e < E) {
        atomicAdd(&relCount[etype[e]], 1);
        __hip_atomic_fetch_add(&cnt[dstA[e]], 1.0f, __ATOMIC_RELAXED, __HIP_MEMORY_SCOPE_AGENT);
    }
}

// prefix-sum over 100 bins + emit (rel, start, count<=CHUNK) chunk descriptors
__global__ void plan(const int* __restrict__ relCount, int* __restrict__ relOff,
                     int* __restrict__ chunks, int* __restrict__ numChunks, int maxChunks) {
    if (threadIdx.x == 0 && blockIdx.x == 0) {
        int run = 0;
        for (int r = 0; r < NREL; ++r) { relOff[r] = run; run += relCount[r]; }
        int nc = 0;
        for (int r = 0; r < NREL; ++r) {
            int c = relCount[r], s = relOff[r];
            while (c > 0 && nc < maxChunks) {
                int take = c < CHUNK ? c : CHUNK;
                chunks[3 * nc] = r; chunks[3 * nc + 1] = s; chunks[3 * nc + 2] = take;
                ++nc; s += take; c -= take;
            }
        }
        *numChunks = nc;
    }
}

__global__ void scatter(const int* __restrict__ etype, const int* __restrict__ relOff,
                        int* __restrict__ cursor, int* __restrict__ perm, int E) {
    int e = blockIdx.x * blockDim.x + threadIdx.x;
    if (e < E) {
        int r = etype[e];
        int pos = atomicAdd(&cursor[r], 1);
        perm[relOff[r] + pos] = e;
    }
}

// ---------------------------------------------------------------------------
// Edge GEMM: per chunk (single relation), TDM-stage Wt_r in LDS, then per
// 16-edge tile run 8(N) x 4(K) WMMA as two interleaved accumulator chains
// (hides WMMA->WMMA RAW hazard NOPs), scale by norm and atomic-scatter.
// ---------------------------------------------------------------------------
__global__ void __launch_bounds__(256) edge_gemm(
        const __bf16* __restrict__ xb, const __bf16* __restrict__ Wt,
        const int* __restrict__ perm, const int* __restrict__ srcA,
        const int* __restrict__ dstA, const float* __restrict__ enorm,
        const int* __restrict__ chunks, const int* __restrict__ numChunks,
        float* __restrict__ agg) {
    __shared__ __align__(32) __bf16 sW[DIM * WSTRIDE];
    __shared__ int   sSrc[8][16];
    __shared__ int   sDst[8][16];
    __shared__ float sNrm[8][16];

    int blk = blockIdx.x;
    if (blk >= *numChunks) return;
    int rel   = chunks[3 * blk];
    int start = chunks[3 * blk + 1];
    int count = chunks[3 * blk + 2];

    const __bf16* Wr = Wt + (size_t)rel * DIM * DIM;
#if HAVE_TDM
    if (threadIdx.x < 32)                      // wave 0 issues the DMA + waits
        tdm_load_tile(Wr, (unsigned)(uintptr_t)&sW[0]);
#else
    coop_load_tile(Wr, sW);
#endif
    __syncthreads();

    const int wave = threadIdx.x >> 5;
    const int lane = threadIdx.x & 31;
    const int row  = lane & 15;
    const int kh   = lane >> 4;                // K-half select (0/1)

    for (int t = 0; t < 2; ++t) {
        int ebase = (wave * 2 + t) * 16;       // edge offset within chunk
        if (lane < 16) {
            int ei = ebase + lane, s = 0, d = 0; float nv = 0.f;
            if (ei < count) {
                int e = perm[start + ei];
                s = srcA[e]; d = dstA[e]; nv = enorm[e];
            }
            sSrc[wave][lane] = s; sDst[wave][lane] = d; sNrm[wave][lane] = nv;
        }
        __builtin_amdgcn_wave_barrier();       // per-wave LDS; DS ops in-order in wave

        // A fragments: 16 edges x 128 K, bf16 row-major (contiguous k = 16*kh + s per lane half)
        const __bf16* xr = xb + (size_t)sSrc[wave][row] * DIM + kh * 16;
        v16bf a0 = *(const v16bf*)(xr +  0);
        v16bf a1 = *(const v16bf*)(xr + 32);
        v16bf a2 = *(const v16bf*)(xr + 64);
        v16bf a3 = *(const v16bf*)(xr + 96);

        unsigned rowIdx[8]; float nA[8];
#pragma unroll
        for (int i = 0; i < 8; ++i) {          // D-matrix row m = i + 8*kh
            rowIdx[i] = (unsigned)sDst[wave][i + kh * 8] << 7;   // 32-bit elem offset
            nA[i]     = sNrm[wave][i + kh * 8];
        }

#pragma unroll
        for (int nt = 0; nt < 8; nt += 2) {    // two independent WMMA chains
            const __bf16* w0 = &sW[((nt    ) * 16 + row) * WSTRIDE + kh * 16];
            const __bf16* w1 = &sW[((nt + 1) * 16 + row) * WSTRIDE + kh * 16];
            v8f c0 = {}, c1 = {};
            c0 = __builtin_amdgcn_wmma_f32_16x16x32_bf16(false, a0, false, *(const v16bf*)(w0 +  0), (short)0, c0, false, false);
            c1 = __builtin_amdgcn_wmma_f32_16x16x32_bf16(false, a0, false, *(const v16bf*)(w1 +  0), (short)0, c1, false, false);
            c0 = __builtin_amdgcn_wmma_f32_16x16x32_bf16(false, a1, false, *(const v16bf*)(w0 + 32), (short)0, c0, false, false);
            c1 = __builtin_amdgcn_wmma_f32_16x16x32_bf16(false, a1, false, *(const v16bf*)(w1 + 32), (short)0, c1, false, false);
            c0 = __builtin_amdgcn_wmma_f32_16x16x32_bf16(false, a2, false, *(const v16bf*)(w0 + 64), (short)0, c0, false, false);
            c1 = __builtin_amdgcn_wmma_f32_16x16x32_bf16(false, a2, false, *(const v16bf*)(w1 + 64), (short)0, c1, false, false);
            c0 = __builtin_amdgcn_wmma_f32_16x16x32_bf16(false, a3, false, *(const v16bf*)(w0 + 96), (short)0, c0, false, false);
            c1 = __builtin_amdgcn_wmma_f32_16x16x32_bf16(false, a3, false, *(const v16bf*)(w1 + 96), (short)0, c1, false, false);
            unsigned col0 = (unsigned)(nt * 16 + row);
            unsigned col1 = col0 + 16u;
#pragma unroll
            for (int i = 0; i < 8; ++i) {      // invalid tail rows carry norm==0
                __hip_atomic_fetch_add(agg + (rowIdx[i] + col0), c0[i] * nA[i],
                                       __ATOMIC_RELAXED, __HIP_MEMORY_SCOPE_AGENT);
                __hip_atomic_fetch_add(agg + (rowIdx[i] + col1), c1[i] * nA[i],
                                       __ATOMIC_RELAXED, __HIP_MEMORY_SCOPE_AGENT);
            }
        }
    }
}

// ---------------------------------------------------------------------------
// Root GEMM + mean-combine + bias (+ReLU -> bf16, else fp32 out), WMMA tiles.
// ---------------------------------------------------------------------------
__global__ void __launch_bounds__(256) node_combine(
        const __bf16* __restrict__ xb, const __bf16* __restrict__ rootT,
        const float* __restrict__ bias, const float* __restrict__ agg,
        const float* __restrict__ cnt, void* __restrict__ outp, int N, int doRelu) {
    __shared__ __align__(32) __bf16 sR[DIM * WSTRIDE];
#if HAVE_TDM
    if (threadIdx.x < 32)
        tdm_load_tile(rootT, (unsigned)(uintptr_t)&sR[0]);
#else
    coop_load_tile(rootT, sR);
#endif
    __syncthreads();

    const int wave = threadIdx.x >> 5;
    const int lane = threadIdx.x & 31;
    const int row  = lane & 15;
    const int kh   = lane >> 4;
    const int base = blockIdx.x * 128 + wave * 16;
    if (base >= N) return;

    int nodeC = base + row; if (nodeC > N - 1) nodeC = N - 1;
    const __bf16* xr = xb + (size_t)nodeC * DIM + kh * 16;
    v16bf a0 = *(const v16bf*)(xr +  0);
    v16bf a1 = *(const v16bf*)(xr + 32);
    v16bf a2 = *(const v16bf*)(xr + 64);
    v16bf a3 = *(const v16bf*)(xr + 96);

    float rcp[8]; int nm[8]; unsigned rowIdx[8];
#pragma unroll
    for (int i = 0; i < 8; ++i) {
        int n = base + i + kh * 8;
        nm[i] = n;
        int ncl = n > N - 1 ? N - 1 : n;
        rowIdx[i] = (unsigned)ncl << 7;
        float c = cnt[ncl];
        rcp[i] = 1.0f / (c > 1.0f ? c : 1.0f);
    }

#pragma unroll
    for (int nt = 0; nt < 8; nt += 2) {
        const __bf16* w0 = &sR[((nt    ) * 16 + row) * WSTRIDE + kh * 16];
        const __bf16* w1 = &sR[((nt + 1) * 16 + row) * WSTRIDE + kh * 16];
        v8f c0 = {}, c1 = {};
        c0 = __builtin_amdgcn_wmma_f32_16x16x32_bf16(false, a0, false, *(const v16bf*)(w0 +  0), (short)0, c0, false, false);
        c1 = __builtin_amdgcn_wmma_f32_16x16x32_bf16(false, a0, false, *(const v16bf*)(w1 +  0), (short)0, c1, false, false);
        c0 = __builtin_amdgcn_wmma_f32_16x16x32_bf16(false, a1, false, *(const v16bf*)(w0 + 32), (short)0, c0, false, false);
        c1 = __builtin_amdgcn_wmma_f32_16x16x32_bf16(false, a1, false, *(const v16bf*)(w1 + 32), (short)0, c1, false, false);
        c0 = __builtin_amdgcn_wmma_f32_16x16x32_bf16(false, a2, false, *(const v16bf*)(w0 + 64), (short)0, c0, false, false);
        c1 = __builtin_amdgcn_wmma_f32_16x16x32_bf16(false, a2, false, *(const v16bf*)(w1 + 64), (short)0, c1, false, false);
        c0 = __builtin_amdgcn_wmma_f32_16x16x32_bf16(false, a3, false, *(const v16bf*)(w0 + 96), (short)0, c0, false, false);
        c1 = __builtin_amdgcn_wmma_f32_16x16x32_bf16(false, a3, false, *(const v16bf*)(w1 + 96), (short)0, c1, false, false);
        unsigned col0 = (unsigned)(nt * 16 + row);
        unsigned col1 = col0 + 16u;
        float b0 = bias[col0], b1 = bias[col1];
#pragma unroll
        for (int i = 0; i < 8; ++i) {
            if (nm[i] < N) {
                unsigned i0 = rowIdx[i] + col0;
                unsigned i1 = rowIdx[i] + col1;
                float v0 = c0[i] + agg[i0] * rcp[i] + b0;
                float v1 = c1[i] + agg[i1] * rcp[i] + b1;
                if (doRelu) {
                    v0 = v0 > 0.f ? v0 : 0.f;
                    v1 = v1 > 0.f ? v1 : 0.f;
                    ((__bf16*)outp)[i0] = (__bf16)v0;
                    ((__bf16*)outp)[i1] = (__bf16)v1;
                } else {
                    ((float*)outp)[i0] = v0;
                    ((float*)outp)[i1] = v1;
                }
            }
        }
    }
}

// ---------------------------------------------------------------------------
extern "C" void kernel_launch(void* const* d_in, const int* in_sizes, int n_in,
                              void* d_out, int out_size, void* d_ws, size_t ws_size,
                              hipStream_t stream) {
    const int N = in_sizes[0];
    const int E = in_sizes[2];

    const int*   entity = (const int*)d_in[0];
    const int*   eidx   = (const int*)d_in[1];
    const int*   etype  = (const int*)d_in[2];
    const float* enorm  = (const float*)d_in[3];
    const float* emb    = (const float*)d_in[4];
    const float* basis1 = (const float*)d_in[5];
    const float* att1   = (const float*)d_in[6];
    const float* root1  = (const float*)d_in[7];
    const float* bias1  = (const float*)d_in[8];
    const float* basis2 = (const float*)d_in[9];
    const float* att2   = (const float*)d_in[10];
    const float* root2  = (const float*)d_in[11];
    const float* bias2  = (const float*)d_in[12];
    const int* srcA = eidx;
    const int* dstA = eidx + E;

    // carve workspace (256B aligned)
    char* ws = (char*)d_ws; size_t off = 0;
    auto take = [&](size_t b) { char* p = ws + off; off = (off + b + 255) & ~(size_t)255; return p; };
    const int maxChunks = E / CHUNK + NREL + 1;
    __bf16* xb  = (__bf16*)take((size_t)N * DIM * 2);
    __bf16* hb  = (__bf16*)take((size_t)N * DIM * 2);
    __bf16* W1t = (__bf16*)take((size_t)NREL * DIM * DIM * 2);
    __bf16* W2t = (__bf16*)take((size_t)NREL * DIM * DIM * 2);
    __bf16* r1t = (__bf16*)take(DIM * DIM * 2);
    __bf16* r2t = (__bf16*)take(DIM * DIM * 2);
    float* agg  = (float*)take((size_t)N * DIM * 4);
    float* cnt  = (float*)take((size_t)N * 4);
    int* perm     = (int*)take((size_t)E * 4);
    int* relCount = (int*)take(NREL * 4);
    int* relOff   = (int*)take((NREL + 1) * 4);
    int* cursor   = (int*)take(NREL * 4);
    int* numCh    = (int*)take(256);
    int* chunks   = (int*)take((size_t)maxChunks * 3 * 4);

    hipMemsetAsync(relCount, 0, NREL * 4, stream);
    hipMemsetAsync(cursor,   0, NREL * 4, stream);
    hipMemsetAsync(cnt,      0, (size_t)N * 4, stream);

    // weight / feature prep
    build_w<<<NREL, 256, 0, stream>>>(basis1, att1, W1t);
    build_w<<<NREL, 256, 0, stream>>>(basis2, att2, W2t);
    build_rt<<<(DIM * DIM + 255) / 256, 256, 0, stream>>>(root1, r1t);
    build_rt<<<(DIM * DIM + 255) / 256, 256, 0, stream>>>(root2, r2t);
    gather_cast<<<((size_t)N * DIM + 255) / 256, 256, 0, stream>>>(entity, emb, xb, N * DIM);

    // relation binning
    hist<<<(E + 255) / 256, 256, 0, stream>>>(etype, dstA, relCount, cnt, E);
    plan<<<1, 32, 0, stream>>>(relCount, relOff, chunks, numCh, maxChunks);
    scatter<<<(E + 255) / 256, 256, 0, stream>>>(etype, relOff, cursor, perm, E);

    // layer 1: edge WMMA + scatter, then root WMMA + mean + bias + ReLU -> hb (bf16)
    hipMemsetAsync(agg, 0, (size_t)N * DIM * 4, stream);
    edge_gemm<<<maxChunks, 256, 0, stream>>>(xb, W1t, perm, srcA, dstA, enorm, chunks, numCh, agg);
    node_combine<<<(N + 127) / 128, 256, 0, stream>>>(xb, r1t, bias1, agg, cnt, hb, N, 1);

    // layer 2: same with hb, fp32 output, no ReLU
    hipMemsetAsync(agg, 0, (size_t)N * DIM * 4, stream);
    edge_gemm<<<maxChunks, 256, 0, stream>>>(hb, W2t, perm, srcA, dstA, enorm, chunks, numCh, agg);
    node_combine<<<(N + 127) / 128, 256, 0, stream>>>(hb, r2t, bias2, agg, cnt, d_out, N, 0);
}
